// GRUBaseline_27032524161059
// MI455X (gfx1250) — matrix operand order repaired and checked by reading, not compiled, not run on hardware
//
#include <hip/hip_runtime.h>

#define BATCH   4096
#define SEQ     200
#define HID     128
#define TILE_M  16
#define NTHREADS 256

typedef __attribute__((ext_vector_type(16))) __bf16 v16bf;
typedef __attribute__((ext_vector_type(8)))  __bf16 v8bf;
typedef __attribute__((ext_vector_type(8)))  float  v8f;

static __device__ __forceinline__ v16bf join8(v8bf lo, v8bf hi) {
    union { v16bf v; v8bf h[2]; } u;
    u.h[0] = lo; u.h[1] = hi;
    return u.v;
}

static __device__ __forceinline__ float sigm(float v) {
    return 1.0f / (1.0f + __expf(-v));
}
static __device__ __forceinline__ float tanh_fast(float v) {
    v = fminf(fmaxf(v, -15.0f), 15.0f);
    float e = __expf(2.0f * v);
    return (e - 1.0f) / (e + 1.0f);
}

// Persistent fused GRU: gather + (emb@Wih^T) + (h@Whh^T) + gates, 200 steps.
// One block = 16 batch rows. Wave w owns hidden columns [16w, 16w+16):
// N-tiles {w, w+8, w+16} = r/z/n gate slices -> gate math is wave-local.
__global__ __launch_bounds__(NTHREADS, 1)
void gru_wmma_27032524161059_kernel(const long long* __restrict__ x,
                                    const float* __restrict__ itemEmb,
                                    const float* __restrict__ Wih,
                                    const float* __restrict__ Whh,
                                    const float* __restrict__ bih,
                                    const float* __restrict__ bhh,
                                    float* __restrict__ out)
{
    __shared__ __align__(32) __bf16 sWih[3 * HID * HID];     // 384x128 bf16 = 96 KB
    __shared__ __align__(32) __bf16 sEmb[2][TILE_M * HID];   // double-buffered, 8 KB
    __shared__ __align__(32) __bf16 sH[2][TILE_M * HID];     // double-buffered, 8 KB

    const int tid  = threadIdx.x;
    const int wave = tid >> 5;     // 0..7
    const int lane = tid & 31;
    const int ln   = lane & 15;    // A row / B column / C column within tile
    const int kh   = lane >> 4;    // K-half selector (WMMA 16-bit layouts)
    const int b0   = blockIdx.x * TILE_M;

    // ---- prologue: W_ih f32 -> bf16 LDS; zero h buffers --------------------
    {
        const float4* w4 = (const float4*)Wih;
        for (int i = tid; i < 3 * HID * HID / 4; i += NTHREADS) {
            float4 f = w4[i];
            __bf16* d = &sWih[i * 4];
            d[0] = (__bf16)f.x; d[1] = (__bf16)f.y;
            d[2] = (__bf16)f.z; d[3] = (__bf16)f.w;
        }
        for (int i = tid; i < TILE_M * HID; i += NTHREADS) {
            sH[0][i] = (__bf16)0.0f;
            sH[1][i] = (__bf16)0.0f;
        }
    }

    // ---- W_hh B-fragments -> VGPRs (recurrent critical path) ---------------
    // B operand (32x16) in this code's contiguous k order: lane ln = column n, kh selects K 0..15 / 16..31;
    // B[k][n] = Whh[n][k], so each lane reads 16 contiguous floats of row n.
    v16bf whh[3][4];
#pragma unroll
    for (int j = 0; j < 3; ++j) {
        const int n = (wave + 8 * j) * 16 + ln;
#pragma unroll
        for (int k = 0; k < 4; ++k) {
            const float* p = Whh + n * HID + k * 32 + kh * 16;
            v16bf f;
#pragma unroll
            for (int e = 0; e < 16; ++e) f[e] = (__bf16)p[e];
            whh[j][k] = f;
        }
    }

    // ---- per-lane gate biases (column cr = wave*16 + ln) -------------------
    const int cr = wave * 16 + ln;
    const float brz = bih[cr] + bhh[cr];                       // r: gx+gh fused
    const float bzz = bih[HID + cr] + bhh[HID + cr];           // z: gx+gh fused
    const float bin = bih[2 * HID + cr];                       // n: kept split
    const float bhn = bhh[2 * HID + cr];

    // ---- gather embeddings for t=0 ----------------------------------------
    {
        const int row = tid >> 4, seg = tid & 15;
        long long idx = x[(long long)(b0 + row) * SEQ];
        const float4* p = (const float4*)(itemEmb + idx * HID + seg * 8);
        float4 a = p[0], b = p[1];
        __bf16* d = &sEmb[0][row * HID + seg * 8];
        d[0]=(__bf16)a.x; d[1]=(__bf16)a.y; d[2]=(__bf16)a.z; d[3]=(__bf16)a.w;
        d[4]=(__bf16)b.x; d[5]=(__bf16)b.y; d[6]=(__bf16)b.z; d[7]=(__bf16)b.w;
    }
    __syncthreads();

    float hreg[8];
#pragma unroll
    for (int r = 0; r < 8; ++r) hreg[r] = 0.0f;

    const int mA = ln;               // A-fragment row (lanes 16-31 repeat rows)
    const int mC = kh * 8;           // C/D row base for this lane half

#pragma unroll 1
    for (int t = 0; t < SEQ; ++t) {
        const int cur = t & 1;
        const int nxt = cur ^ 1;

        // issue next step's gather early (hides global latency behind WMMAs)
        if (t + 1 < SEQ) {
            const int row = tid >> 4, seg = tid & 15;
            long long idx = x[(long long)(b0 + row) * SEQ + (t + 1)];
            const float4* p = (const float4*)(itemEmb + idx * HID + seg * 8);
            float4 a = p[0], b = p[1];
            __bf16* d = &sEmb[nxt][row * HID + seg * 8];
            d[0]=(__bf16)a.x; d[1]=(__bf16)a.y; d[2]=(__bf16)a.z; d[3]=(__bf16)a.w;
            d[4]=(__bf16)b.x; d[5]=(__bf16)b.y; d[6]=(__bf16)b.z; d[7]=(__bf16)b.w;
        }

        // ---- dual GEMM: gx (emb x Wih^T from LDS), gh (h x Whh^T from regs)
        v8f gx0 = {0,0,0,0,0,0,0,0}, gx1 = {0,0,0,0,0,0,0,0}, gx2 = {0,0,0,0,0,0,0,0};
        v8f gh0 = {0,0,0,0,0,0,0,0}, gh1 = {0,0,0,0,0,0,0,0}, gh2 = {0,0,0,0,0,0,0,0};
#pragma unroll
        for (int k = 0; k < 4; ++k) {
            // A operand (16x32) in this code's contiguous k order: per lane 16 contiguous elements:
            // K = k*32 + 16*kh .. +15 (matches B)
            const __bf16* er = &sEmb[cur][mA * HID + k * 32 + kh * 16];
            v16bf aE = join8(*(const v8bf*)er, *(const v8bf*)(er + 8));
            const __bf16* hr = &sH[cur][mA * HID + k * 32 + kh * 16];
            v16bf aH = join8(*(const v8bf*)hr, *(const v8bf*)(hr + 8));

            const __bf16* bR = &sWih[((wave     ) * 16 + ln) * HID + k * 32 + kh * 16];
            const __bf16* bZ = &sWih[((wave +  8) * 16 + ln) * HID + k * 32 + kh * 16];
            const __bf16* bN = &sWih[((wave + 16) * 16 + ln) * HID + k * 32 + kh * 16];

            gx0 = __builtin_amdgcn_wmma_f32_16x16x32_bf16(false, aE, false, *(const v16bf*)bR, (short)0, gx0, false, false);
            gh0 = __builtin_amdgcn_wmma_f32_16x16x32_bf16(false, aH, false, whh[0][k],        (short)0, gh0, false, false);
            gx1 = __builtin_amdgcn_wmma_f32_16x16x32_bf16(false, aE, false, *(const v16bf*)bZ, (short)0, gx1, false, false);
            gh1 = __builtin_amdgcn_wmma_f32_16x16x32_bf16(false, aH, false, whh[1][k],        (short)0, gh1, false, false);
            gx2 = __builtin_amdgcn_wmma_f32_16x16x32_bf16(false, aE, false, *(const v16bf*)bN, (short)0, gx2, false, false);
            gh2 = __builtin_amdgcn_wmma_f32_16x16x32_bf16(false, aH, false, whh[2][k],        (short)0, gh2, false, false);
        }

        // ---- wave-local gate math (fp32), h kept in C/D register layout ----
#pragma unroll
        for (int r = 0; r < 8; ++r) {
            float rg = sigm(gx0[r] + gh0[r] + brz);
            float zg = sigm(gx1[r] + gh1[r] + bzz);
            float ng = tanh_fast(gx2[r] + bin + rg * (gh2[r] + bhn));
            float hn = (1.0f - zg) * ng + zg * hreg[r];
            hreg[r] = hn;
            sH[nxt][(mC + r) * HID + cr] = (__bf16)hn;   // bf16 copy for next A
        }
        __syncthreads();   // single barrier/step thanks to double buffering
    }

    // ---- final hidden state -> d_out [4096,128] f32 ------------------------
#pragma unroll
    for (int r = 0; r < 8; ++r) {
        out[(long long)(b0 + mC + r) * HID + cr] = hreg[r];
    }
}

extern "C" void kernel_launch(void* const* d_in, const int* in_sizes, int n_in,
                              void* d_out, int out_size, void* d_ws, size_t ws_size,
                              hipStream_t stream) {
    (void)in_sizes; (void)n_in; (void)out_size; (void)d_ws; (void)ws_size;
    const long long* x   = (const long long*)d_in[0];   // int64 [4096,200]
    const float* itemEmb = (const float*)d_in[1];       // [100000,128]
    const float* Wih     = (const float*)d_in[2];       // [384,128]
    const float* Whh     = (const float*)d_in[3];       // [384,128]
    const float* bih     = (const float*)d_in[4];       // [384]
    const float* bhh     = (const float*)d_in[5];       // [384]
    float* out           = (float*)d_out;               // [4096,128]

    dim3 grid(BATCH / TILE_M);   // 256 blocks
    dim3 block(NTHREADS);        // 8 waves (wave32)
    gru_wmma_27032524161059_kernel<<<grid, block, 0, stream>>>(
        x, itemEmb, Wih, Whh, bih, bhh, out);
}
